// MultiHeadAttentionBlock_22016002359767
// MI455X (gfx1250) — matrix operand order repaired and checked by reading, not compiled, not run on hardware
//
#include <hip/hip_runtime.h>

// ---------------------------------------------------------------------------
// MultiHeadAttentionBlock for MI455X (gfx1250, wave32, WMMA bf16)
//   out = softmax_causal( (xWq^T+bq)(xWk^T+bk)^T / sqrt(dk) ) (xWv^T+bv) Wo^T + bo
// Strategy: one memory-bound f32->bf16 pass for activations + weights, then
// every contraction is a pure load/v_wmma_f32_16x16x32_bf16 loop.
// ---------------------------------------------------------------------------

#define D_MODEL 1024
#define NHEAD   16
#define DKH     64
#define BATCH   2
#define SEQ     2048
#define MTOT    (BATCH * SEQ)   // 4096 rows in the projection GEMMs

typedef __attribute__((ext_vector_type(16))) __bf16          v16bf;
typedef __attribute__((ext_vector_type(8)))  float           v8f;
typedef __attribute__((ext_vector_type(8)))  unsigned short  v8us;

union Frag {                 // 32 bytes: one WMMA A/B operand per lane
    v16bf bf;
    v8us  us[2];
};

__device__ __forceinline__ unsigned short f2bf_bits(float f) {
    unsigned u = __builtin_bit_cast(unsigned, f);
    unsigned r = u + 0x7FFFu + ((u >> 16) & 1u);   // round-to-nearest-even
    return (unsigned short)(r >> 16);
}
__device__ __forceinline__ __bf16 bits2bf(unsigned short h) {
    return __builtin_bit_cast(__bf16, h);
}

// ---------------------------------------------------------------------------
// Elementwise f32 -> bf16 (memory bound, runs once; ~4us of HBM traffic).
// ---------------------------------------------------------------------------
__global__ __launch_bounds__(256)
void cvt_f32_bf16_kernel(const float* __restrict__ src,
                         unsigned short* __restrict__ dst, int n)
{
    const int idx = (blockIdx.x * 256 + threadIdx.x) * 8;
    if (idx >= n) return;
#pragma unroll
    for (int j = 0; j < 8; ++j) dst[idx + j] = f2bf_bits(src[idx + j]);
}

// ---------------------------------------------------------------------------
// Tiled GEMM:  Out[m][n] = sum_k A[m][k] * W[n][k]  (+ bias[n]), all-bf16 in.
//   8 waves/block, wave tile 32x64 (2x4 WMMA tiles), block tile 128x128.
// A fragment = ISA 16-bit A layout (lane<16: K 0..7,16..23 ; lane>=16: +8)
//   -> two global_load_b128 per tile per lane.
// B fragment = one W row (a column of W^T) per lane, same K order as A
//   -> two global_load_b128 per tile per lane.
// ---------------------------------------------------------------------------
template <bool VT_STORE, bool OUT_F32>
__global__ __launch_bounds__(256)
void wmma_gemm_kernel(const unsigned short* __restrict__ A,
                      const unsigned short* __restrict__ W,
                      const float* __restrict__ bias,
                      void* __restrict__ Outv)
{
    const int lane = threadIdx.x & 31;
    const int wave = threadIdx.x >> 5;
    const int half = lane >> 4;
    const int r    = lane & 15;

    const int wm    = wave & 3;                       // 4 waves along M
    const int wn    = wave >> 2;                      // 2 waves along N
    const int mBase = blockIdx.y * 128 + wm * 32;
    const int nBase = blockIdx.x * 128 + wn * 64;

    v8f acc[2][4];
#pragma unroll
    for (int mi = 0; mi < 2; ++mi)
#pragma unroll
        for (int ni = 0; ni < 4; ++ni)
#pragma unroll
            for (int i = 0; i < 8; ++i) acc[mi][ni][i] = 0.0f;

    const unsigned short* aRow0 = A + (size_t)(mBase + r) * D_MODEL + half * 8;
    const unsigned short* bRow0 = W + (size_t)(nBase + r) * D_MODEL + half * 8;

    for (int k0 = 0; k0 < D_MODEL; k0 += 32) {
        Frag a[2], b[4];
#pragma unroll
        for (int mi = 0; mi < 2; ++mi) {
            const unsigned short* p = aRow0 + (size_t)mi * 16 * D_MODEL + k0;
            a[mi].us[0] = *(const v8us*)(p);        // K 0..7  (this half's base)
            a[mi].us[1] = *(const v8us*)(p + 16);   // K 16..23
        }
#pragma unroll
        for (int ni = 0; ni < 4; ++ni) {
            const unsigned short* p = bRow0 + (size_t)ni * 16 * D_MODEL + k0;
            b[ni].us[0] = *(const v8us*)(p);        // K 0..7  (this half's base)
            b[ni].us[1] = *(const v8us*)(p + 16);
        }
#pragma unroll
        for (int mi = 0; mi < 2; ++mi)
#pragma unroll
            for (int ni = 0; ni < 4; ++ni)
                acc[mi][ni] = __builtin_amdgcn_wmma_f32_16x16x32_bf16(
                    false, a[mi].bf, false, b[ni].bf, (short)0, acc[mi][ni],
                    false, false);
    }

    // Epilogue: C layout -> VGPR i holds row (i + 8*half), col = lane&15.
#pragma unroll
    for (int mi = 0; mi < 2; ++mi) {
#pragma unroll
        for (int ni = 0; ni < 4; ++ni) {
#pragma unroll
            for (int i = 0; i < 8; ++i) {
                const int   row = mBase + mi * 16 + i + 8 * half;
                const int   col = nBase + ni * 16 + r;
                const float v   = acc[mi][ni][i] + bias[col];
                if constexpr (OUT_F32) {
                    ((float*)Outv)[(size_t)row * D_MODEL + col] = v;
                } else if constexpr (VT_STORE) {
                    // store V transposed per head: [b, h, dk, s] for the PV GEMM
                    const int b_  = row >> 11;       // row / SEQ
                    const int s_  = row & (SEQ - 1);
                    const int h_  = col >> 6;        // col / DKH
                    const int dk_ = col & (DKH - 1);
                    ((unsigned short*)Outv)[(((size_t)(b_ * NHEAD + h_) * DKH + dk_) << 11) + s_] =
                        f2bf_bits(v);
                } else {
                    ((unsigned short*)Outv)[(size_t)row * D_MODEL + col] = f2bf_bits(v);
                }
            }
        }
    }
}

// ---------------------------------------------------------------------------
// Causal flash attention, one (b,h,128-query block) per workgroup.
// Each wave owns 16 query rows: online softmax over 32-key blocks.
//   scores: 4x v_wmma (16x16x32, dk contraction of 64)
//   PV:     4x v_wmma (key contraction of 32), P staged through per-wave LDS.
// ---------------------------------------------------------------------------
__global__ __launch_bounds__(256)
void flash_attn_kernel(const unsigned short* __restrict__ Qb,
                       const unsigned short* __restrict__ Kb,
                       const unsigned short* __restrict__ Vt,
                       unsigned short* __restrict__ Ctx)
{
    __shared__ unsigned short pbuf[8][16 * 32];   // per-wave P staging (8 KB)

    const int lane = threadIdx.x & 31;
    const int wave = threadIdx.x >> 5;
    const int half = lane >> 4;
    const int r    = lane & 15;

    const int bh = blockIdx.y;
    const int b_ = bh >> 4;
    const int h_ = bh & 15;
    const int q0 = blockIdx.x * 128 + wave * 16;   // first query row of this wave

    const size_t qkBase = (size_t)b_ * SEQ * D_MODEL + (size_t)h_ * DKH;
    const unsigned short* Qp = Qb + qkBase;
    const unsigned short* Kp = Kb + qkBase;
    const unsigned short* Vp = Vt + ((size_t)(b_ * NHEAD + h_) * DKH << 11);
    unsigned short*       Cp = Ctx + qkBase;

    // Q fragments (A layout), resident for the whole key loop.
    Frag qf[2];
#pragma unroll
    for (int c = 0; c < 2; ++c) {
        const unsigned short* p = Qp + (size_t)(q0 + r) * D_MODEL + c * 32 + half * 8;
        qf[c].us[0] = *(const v8us*)(p);
        qf[c].us[1] = *(const v8us*)(p + 16);
    }

    v8f o[4];
#pragma unroll
    for (int t = 0; t < 4; ++t)
#pragma unroll
        for (int i = 0; i < 8; ++i) o[t][i] = 0.0f;

    float m_i[8], l_i[8];
#pragma unroll
    for (int i = 0; i < 8; ++i) { m_i[i] = -1e30f; l_i[i] = 0.0f; }

    const int nkb = (q0 + 47) >> 5;   // causal: keys 0 .. q0+15
    for (int kb = 0; kb < nkb; ++kb) {
        const int k0 = kb * 32;

        // --- scores: S = Q K^T / sqrt(dk), two 16-key tiles ---
        v8f s[2];
#pragma unroll
        for (int t = 0; t < 2; ++t) {
#pragma unroll
            for (int i = 0; i < 8; ++i) s[t][i] = 0.0f;
#pragma unroll
            for (int c = 0; c < 2; ++c) {
                Frag kf;
                const unsigned short* p =
                    Kp + (size_t)(k0 + t * 16 + r) * D_MODEL + c * 32 + half * 8;
                kf.us[0] = *(const v8us*)(p);
                kf.us[1] = *(const v8us*)(p + 16);
                s[t] = __builtin_amdgcn_wmma_f32_16x16x32_bf16(
                    false, qf[c].bf, false, kf.bf, (short)0, s[t], false, false);
            }
        }

        // --- scale + causal mask + online softmax (half-wave row reductions) ---
#pragma unroll
        for (int i = 0; i < 8; ++i) {
            const int row = q0 + i + 8 * half;
#pragma unroll
            for (int t = 0; t < 2; ++t) {
                const int col = k0 + t * 16 + r;
                float v = s[t][i] * 0.125f;          // 1/sqrt(64)
                if (col > row) v = -1e30f;
                s[t][i] = v;
            }
            float mb = fmaxf(s[0][i], s[1][i]);
#pragma unroll
            for (int mk = 1; mk < 16; mk <<= 1)
                mb = fmaxf(mb, __shfl_xor(mb, mk, 32));  // stays within lane half

            const float mn   = fmaxf(m_i[i], mb);
            const float corr = __expf(m_i[i] - mn);
            m_i[i] = mn;
            l_i[i] *= corr;
#pragma unroll
            for (int t = 0; t < 4; ++t) o[t][i] *= corr;

            float rs = 0.0f;
#pragma unroll
            for (int t = 0; t < 2; ++t) {
                const float pv = __expf(s[t][i] - mn);
                s[t][i] = pv;
                rs += pv;
            }
#pragma unroll
            for (int mk = 1; mk < 16; mk <<= 1)
                rs += __shfl_xor(rs, mk, 32);
            l_i[i] += rs;

            // stage P (C layout) into this wave's LDS slice as bf16
#pragma unroll
            for (int t = 0; t < 2; ++t)
                pbuf[wave][(i + 8 * half) * 32 + t * 16 + r] = f2bf_bits(s[t][i]);
        }

        // --- reload P in A layout (same-wave LDS ops are in order) ---
        Frag pf;
        {
            const unsigned short* p = &pbuf[wave][r * 32 + half * 8];
            pf.us[0] = *(const v8us*)(p);        // 16B aligned: (r*32+half*8)*2
            pf.us[1] = *(const v8us*)(p + 16);
        }

        // --- O += P V, V^T layout gives contiguous key runs per lane ---
#pragma unroll
        for (int t = 0; t < 4; ++t) {
            Frag vf;
            const unsigned short* p = Vp + ((size_t)(t * 16 + r) << 11) + k0 + half * 8;
            vf.us[0] = *(const v8us*)(p);
            vf.us[1] = *(const v8us*)(p + 16);
            o[t] = __builtin_amdgcn_wmma_f32_16x16x32_bf16(
                false, pf.bf, false, vf.bf, (short)0, o[t], false, false);
        }
    }

    // --- normalize and emit context (bf16, [b, s, h*dk] layout) ---
#pragma unroll
    for (int i = 0; i < 8; ++i) {
        const float inv = 1.0f / l_i[i];
        const int   row = q0 + i + 8 * half;
#pragma unroll
        for (int t = 0; t < 4; ++t)
            Cp[(size_t)row * D_MODEL + t * 16 + r] = f2bf_bits(o[t][i] * inv);
    }
}

// ---------------------------------------------------------------------------
// Launch: f32->bf16 conversions -> 3 projection GEMMs -> attention -> out GEMM.
// Workspace (bf16): qbf|kbf|vbf (3x8MB) wq|wk|wv|wo (4x2MB) Q|K|Vt|Ctx (4x8MB)
//   = 64 MB total.
// ---------------------------------------------------------------------------
extern "C" void kernel_launch(void* const* d_in, const int* in_sizes, int n_in,
                              void* d_out, int out_size, void* d_ws, size_t ws_size,
                              hipStream_t stream)
{
    const float* q  = (const float*)d_in[0];
    const float* k  = (const float*)d_in[1];
    const float* v  = (const float*)d_in[2];
    // d_in[3] = causal mask (implicit in the kernel)
    const float* Wq = (const float*)d_in[4];
    const float* bq = (const float*)d_in[5];
    const float* Wk = (const float*)d_in[6];
    const float* bk = (const float*)d_in[7];
    const float* Wv = (const float*)d_in[8];
    const float* bv = (const float*)d_in[9];
    const float* Wo = (const float*)d_in[10];
    const float* bo = (const float*)d_in[11];

    const size_t actElems = (size_t)MTOT * D_MODEL;     // 4M elements
    const size_t wElems   = (size_t)D_MODEL * D_MODEL;  // 1M elements

    unsigned short* p   = (unsigned short*)d_ws;
    unsigned short* qbf = p; p += actElems;
    unsigned short* kbf = p; p += actElems;
    unsigned short* vbf = p; p += actElems;
    unsigned short* wqb = p; p += wElems;
    unsigned short* wkb = p; p += wElems;
    unsigned short* wvb = p; p += wElems;
    unsigned short* wob = p; p += wElems;
    unsigned short* Qbf = p; p += actElems;
    unsigned short* Kbf = p; p += actElems;
    unsigned short* Vt  = p; p += actElems;
    unsigned short* Ctx = p; p += actElems;

    dim3 blk(256);
    const int cvtA = (int)(actElems / 8 / 256);   // 2048 blocks
    const int cvtW = (int)(wElems   / 8 / 256);   // 512 blocks

    cvt_f32_bf16_kernel<<<cvtA, blk, 0, stream>>>(q,  qbf, (int)actElems);
    cvt_f32_bf16_kernel<<<cvtA, blk, 0, stream>>>(k,  kbf, (int)actElems);
    cvt_f32_bf16_kernel<<<cvtA, blk, 0, stream>>>(v,  vbf, (int)actElems);
    cvt_f32_bf16_kernel<<<cvtW, blk, 0, stream>>>(Wq, wqb, (int)wElems);
    cvt_f32_bf16_kernel<<<cvtW, blk, 0, stream>>>(Wk, wkb, (int)wElems);
    cvt_f32_bf16_kernel<<<cvtW, blk, 0, stream>>>(Wv, wvb, (int)wElems);
    cvt_f32_bf16_kernel<<<cvtW, blk, 0, stream>>>(Wo, wob, (int)wElems);

    dim3 gg(D_MODEL / 128, MTOT / 128);     // (8, 32)
    dim3 ga(SEQ / 128, BATCH * NHEAD);      // (16, 32)

    wmma_gemm_kernel<false, false><<<gg, blk, 0, stream>>>(qbf, wqb, bq, Qbf);
    wmma_gemm_kernel<false, false><<<gg, blk, 0, stream>>>(kbf, wkb, bk, Kbf);
    wmma_gemm_kernel<true,  false><<<gg, blk, 0, stream>>>(vbf, wvb, bv, Vt);
    flash_attn_kernel<<<ga, blk, 0, stream>>>(Qbf, Kbf, Vt, Ctx);
    wmma_gemm_kernel<false, true ><<<gg, blk, 0, stream>>>(Ctx, wob, bo, d_out);
}
